// AlignerModel_42838003810924
// MI455X (gfx1250) — matrix operand order repaired and checked by reading, not compiled, not run on hardware
//
#include <hip/hip_runtime.h>

// ---------------------------------------------------------------------------
// MI455X (gfx1250) aligner model.
//  - All GEMM-shaped work (5 convs + f.t distance einsum) on
//    v_wmma_f32_16x16x32_f16 with f32 accumulation.
//  - Intermediate activations stored f16 (halves traffic, removes cvt from
//    inner loops); everything stays L2-resident (192MB L2 >> 35MB working set).
//  - Score GEMM stages its B (t-matrix) tile into LDS via the Tensor Data
//    Mover (tensor_load_to_lds + s_wait_tensorcnt), B-frags read via ds_load.
//  - Sequential DPs (CTC forward-sum, MAS) use LDS double buffers, one
//    workgroup per batch; all reductions fixed-order (deterministic).
// ---------------------------------------------------------------------------

typedef __attribute__((ext_vector_type(16))) _Float16 v16h;
typedef __attribute__((ext_vector_type(8)))  float    v8f;
typedef __attribute__((ext_vector_type(4)))  unsigned int u32x4;
typedef __attribute__((ext_vector_type(8)))  int      i32x8;
typedef __attribute__((ext_vector_type(4)))  int      i32x4;

#define BB      16
#define TFEATS  2000
#define TTEXT   400
#define NADIM   256
#define NODIM   512
#define BIG_NEG (-1e30f)
#define BLANKLP (-1.0f)
#define S_CTC   (2 * TTEXT + 1)   // 801

static __device__ __forceinline__ float lae(float a, float b) {
    float m = fmaxf(a, b);
    float n = fminf(a, b);
    if (m <= -1e29f) return m;
    return m + log1pf(__expf(n - m));
}

// ------------------------- fragment helpers --------------------------------
// A fragment (16x32 f16), this code's contiguous-K order: lane L (hs=L>>4, m=L&15):
//   half h -> K = hs*16 + h (matches B fragment)
static __device__ __forceinline__ v16h zero_frag() {
    v16h a;
#pragma unroll
    for (int i = 0; i < 16; ++i) a[i] = (_Float16)0.0f;
    return a;
}

template <typename SRC>
static __device__ __forceinline__ v16h load_a_frag(const SRC* xrow, int cbase, int hs) {
    const SRC* p = xrow + cbase + hs * 16;
    v16h a;
#pragma unroll
    for (int i = 0; i < 16; ++i) a[i] = (_Float16)p[i];
    return a;
}

// B fragment (32x16 f16): lane L holds column N=L&15, half h -> K = (L>>4)*16 + h.
// Source is N-major (row n of src = column n of B), contiguous in K.
static __device__ __forceinline__ v16h load_b_frag(const _Float16* src, int Kld,
                                                   int n, int jbase, int hs) {
    const _Float16* p = src + (size_t)n * Kld + jbase + hs * 16;
    v16h b;
#pragma unroll
    for (int i = 0; i < 16; ++i) b[i] = p[i];
    return b;
}

// ------------------------- weight repack -----------------------------------
// w[o][c][k] (f32) -> wt[o][k*C + c] (f16): contiguous B-frag reads.
__global__ void aligner_repack_w(const float* __restrict__ w, _Float16* __restrict__ out,
                                 int O, int C, int Kw) {
    int idx = blockIdx.x * blockDim.x + threadIdx.x;
    int total = O * C * Kw;
    if (idx >= total) return;
    int o = idx / (C * Kw);
    int r = idx - o * (C * Kw);
    int kk = r / C;
    int c  = r - kk * C;
    out[idx] = (_Float16)w[(size_t)o * C * Kw + (size_t)c * Kw + kk];
}

// ------------------------- beta-binomial prior -----------------------------
__global__ void aligner_prior(float* __restrict__ prior) {
    int idx = blockIdx.x * blockDim.x + threadIdx.x;
    if (idx >= TFEATS * TTEXT) return;
    int tt = idx / TTEXT;
    int k  = idx - tt * TTEXT;
    float t  = (float)(tt + 1);
    float a  = t;
    float bb = (float)TFEATS - t + 1.0f;
    float kk = (float)(k + 1);
    float n  = (float)TTEXT;
    float logC = lgammaf(n + 1.0f) - lgammaf(kk + 1.0f) - lgammaf(n - kk + 1.0f);
    float bl1 = lgammaf(kk + a) + lgammaf(n - kk + bb) - lgammaf(n + a + bb);
    float bl0 = lgammaf(a) + lgammaf(bb) - lgammaf(a + bb);
    prior[idx] = logC + bl1 - bl0;
}

// ------------------------- conv1d as implicit-im2col WMMA GEMM -------------
// y[b][t][o] = bias[o] + sum_{k,c} x[b][t+k-pad][c] * wt[o][k*CIN+c] (opt ReLU)
// One wave per block; each wave: 16 (time) x 64 (cout) via 4 accumulators.
// grid = (T/16, 256/64, B), block = 32.
template <int CIN, int KS, typename SRC>
__global__ void aligner_conv_gemm(const SRC* __restrict__ x, const _Float16* __restrict__ wt,
                                  const float* __restrict__ bias, _Float16* __restrict__ y,
                                  int T, int relu) {
    constexpr int KD = CIN * KS;
    constexpr int PAD = KS >> 1;
    constexpr int COUT = 256;
    const int lane = threadIdx.x & 31;
    const int hs = lane >> 4;
    const int m  = lane & 15;
    const int b = blockIdx.z;
    const int mbase = blockIdx.x * 16;
    const int nbase0 = blockIdx.y * 64;
    const SRC* xb = x + (size_t)b * T * CIN;

    v8f acc[4];
#pragma unroll
    for (int nb = 0; nb < 4; ++nb)
#pragma unroll
        for (int r = 0; r < 8; ++r) acc[nb][r] = 0.f;

#pragma unroll
    for (int kk = 0; kk < KS; ++kk) {
        const int tt = mbase + m + kk - PAD;
        const bool valid = (tt >= 0) && (tt < T);
        const SRC* xrow = xb + (size_t)(valid ? tt : 0) * CIN;
        for (int cb = 0; cb < CIN; cb += 32) {
            v16h af = valid ? load_a_frag<SRC>(xrow, cb, hs) : zero_frag();
#pragma unroll
            for (int nb = 0; nb < 4; ++nb) {
                v16h bf = load_b_frag(wt, KD, nbase0 + nb * 16 + m, kk * CIN + cb, hs);
                acc[nb] = __builtin_amdgcn_wmma_f32_16x16x32_f16(
                    false, af, false, bf, (short)0, acc[nb], false, false);
            }
        }
    }

    _Float16* yb = y + (size_t)b * T * COUT;
#pragma unroll
    for (int nb = 0; nb < 4; ++nb) {
        const int n = nbase0 + nb * 16 + m;
        const float bv = bias[n];
#pragma unroll
        for (int r = 0; r < 8; ++r) {
            int mm = mbase + r + 8 * hs;
            float v = acc[nb][r] + bv;
            if (relu) v = fmaxf(v, 0.f);
            yb[(size_t)mm * COUT + n] = (_Float16)v;
        }
    }
}

// ------------------------- row squared-norms (f16 in, f32 out) -------------
__global__ void aligner_row_norm(const _Float16* __restrict__ x, float* __restrict__ out,
                                 int rows, int K) {
    int r = blockIdx.x * blockDim.x + threadIdx.x;
    if (r >= rows) return;
    const _Float16* p = x + (size_t)r * K;
    float s = 0.f;
    for (int i = 0; i < K; ++i) { float v = (float)p[i]; s += v * v; }
    out[r] = s;
}

// ------------------------- distance-score GEMM (f.t) -----------------------
// score[b][ts][tx] = -sqrt(max(|f|^2+|t|^2-2 f.t, 1e-12))
// One wave per block: 16 (ts) x 80 (tx) via 5 accumulators (400 = 5*80).
// B tile (80 x 256 f16 rows of t) is DMA'd into LDS by the Tensor Data Mover.
// grid = (2000/16, 400/80, B), block = 32.
__global__ void aligner_score_gemm(const _Float16* __restrict__ f, const _Float16* __restrict__ t,
                                   const float* __restrict__ fn, const float* __restrict__ tn,
                                   float* __restrict__ out) {
    __shared__ _Float16 bsm[80 * NADIM];          // 40 KB
    const int lane = threadIdx.x & 31;
    const int hs = lane >> 4;
    const int m  = lane & 15;
    const int b = blockIdx.z;
    const int mbase = blockIdx.x * 16;            // ts
    const int nbase0 = blockIdx.y * 80;           // tx
    const _Float16* fb = f + (size_t)b * TFEATS * NADIM;
    const _Float16* tb = t + ((size_t)b * TTEXT + nbase0) * NADIM;

    // --- TDM: 2D tile load, data_size=2B, tensor 256 x 400, tile 256 x 80 ---
    {
        unsigned lds_off = (unsigned)(size_t)(&bsm[0]);          // LDS byte offset
        unsigned long long ga = (unsigned long long)(size_t)tb;  // tile start
        u32x4 g0;
        g0[0] = 1u;                                              // count=1 (valid user D#)
        g0[1] = lds_off;                                         // lds_addr
        g0[2] = (unsigned)(ga & 0xFFFFFFFFu);                    // global_addr[31:0]
        g0[3] = (unsigned)((ga >> 32) & 0x01FFFFFFu) | (2u << 30); // addr[56:32] | type=2
        i32x8 g1;
        g1[0] = (int)(1u << 16);          // data_size=1 (2 bytes), no multicast/pad
        g1[1] = (int)(((unsigned)NADIM) << 16);   // tensor_dim0[15:0] in bits[31:16]
        g1[2] = (int)(((unsigned)TTEXT) << 16);   // dim0[31:16]=0 | tensor_dim1[15:0]
        g1[3] = (int)(((unsigned)NADIM) << 16);   // dim1[31:16]=0 | tile_dim0=256
        g1[4] = 80;                       // tile_dim1=80, tile_dim2=0
        g1[5] = NADIM;                    // tensor_dim0_stride[31:0] = 256
        g1[6] = 0;
        g1[7] = 0;
        i32x4 gz; gz[0] = 0; gz[1] = 0; gz[2] = 0; gz[3] = 0;
#if defined(__clang_major__) && (__clang_major__ >= 23)
        i32x8 gz8;
#pragma unroll
        for (int i = 0; i < 8; ++i) gz8[i] = 0;
        __builtin_amdgcn_tensor_load_to_lds(g0, g1, gz, gz, gz8, 0);
#else
        __builtin_amdgcn_tensor_load_to_lds(g0, g1, gz, gz, 0);
#endif
        __builtin_amdgcn_s_wait_tensorcnt(0);
    }

    v8f acc[5];
#pragma unroll
    for (int nb = 0; nb < 5; ++nb)
#pragma unroll
        for (int r = 0; r < 8; ++r) acc[nb][r] = 0.f;

    const _Float16* frow = fb + (size_t)(mbase + m) * NADIM;
    for (int j = 0; j < NADIM; j += 32) {
        if (j + 32 < NADIM) __builtin_prefetch(frow + j + 32, 0, 3);
        v16h af = load_a_frag<_Float16>(frow, j, hs);
#pragma unroll
        for (int nb = 0; nb < 5; ++nb) {
            v16h bf = load_b_frag(bsm, NADIM, nb * 16 + m, j, hs);   // ds_load
            acc[nb] = __builtin_amdgcn_wmma_f32_16x16x32_f16(
                false, af, false, bf, (short)0, acc[nb], false, false);
        }
    }

    float* ob = out + (size_t)b * TFEATS * TTEXT;
#pragma unroll
    for (int nb = 0; nb < 5; ++nb) {
        const int n = nbase0 + nb * 16 + m;
        const float tnv = tn[b * TTEXT + n];
#pragma unroll
        for (int r = 0; r < 8; ++r) {
            int mm = mbase + r + 8 * hs;
            float d2 = fn[b * TFEATS + mm] + tnv - 2.0f * acc[nb][r];
            ob[(size_t)mm * TTEXT + n] = -sqrtf(fmaxf(d2, 1e-12f));
        }
    }
}

// ------------------------- log_softmax over tx + prior (in place) ----------
__global__ void aligner_softmax_prior(float* __restrict__ lp, const float* __restrict__ prior) {
    int gwave = (blockIdx.x * blockDim.x + threadIdx.x) >> 5;
    int lane = threadIdx.x & 31;
    if (gwave >= BB * TFEATS) return;
    int ts = gwave % TFEATS;
    float* row = lp + (size_t)gwave * TTEXT;

    float v[13];
    float mx = BIG_NEG;
#pragma unroll
    for (int i = 0; i < 13; ++i) {
        int j = lane + i * 32;
        v[i] = (j < TTEXT) ? row[j] : BIG_NEG;
        mx = fmaxf(mx, v[i]);
    }
    for (int off = 16; off > 0; off >>= 1) mx = fmaxf(mx, __shfl_xor(mx, off, 32));
    float s = 0.f;
#pragma unroll
    for (int i = 0; i < 13; ++i) {
        int j = lane + i * 32;
        if (j < TTEXT) s += __expf(v[i] - mx);
    }
    for (int off = 16; off > 0; off >>= 1) s += __shfl_xor(s, off, 32);
    float lse = mx + __logf(s);
#pragma unroll
    for (int i = 0; i < 13; ++i) {
        int j = lane + i * 32;
        if (j < TTEXT) row[j] = v[i] - lse + prior[(size_t)ts * TTEXT + j];
    }
}

// ------------------------- CTC forward-sum ---------------------------------
__global__ void aligner_ctc(const float* __restrict__ lp, float* __restrict__ nll) {
    const int b = blockIdx.x;
    const int tid = threadIdx.x;
    const int lane = tid & 31;
    const int wv = tid >> 5;
    const float* lpb = lp + (size_t)b * TFEATS * TTEXT;

    __shared__ float row[TTEXT + 1];
    __shared__ float aA[S_CTC], aBuf[S_CTC];
    __shared__ float red[16];
    __shared__ float s_mx, s_lse;

    float* aold = aA;
    float* anew = aBuf;

    for (int t = 0; t < TFEATS; ++t) {
        for (int j = tid; j < TTEXT + 1; j += blockDim.x)
            row[j] = (j == 0) ? BLANKLP : lpb[(size_t)t * TTEXT + (j - 1)];
        __syncthreads();
        float lm = BIG_NEG;
        for (int j = tid; j < TTEXT + 1; j += blockDim.x) lm = fmaxf(lm, row[j]);
        for (int off = 16; off > 0; off >>= 1) lm = fmaxf(lm, __shfl_xor(lm, off, 32));
        if (lane == 0) red[wv] = lm;
        __syncthreads();
        if (tid == 0) {
            float m = red[0];
            for (int i = 1; i < 16; ++i) m = fmaxf(m, red[i]);
            s_mx = m;
        }
        __syncthreads();
        float ls = 0.f;
        for (int j = tid; j < TTEXT + 1; j += blockDim.x) ls += __expf(row[j] - s_mx);
        for (int off = 16; off > 0; off >>= 1) ls += __shfl_xor(ls, off, 32);
        if (lane == 0) red[wv] = ls;
        __syncthreads();
        if (tid == 0) {
            float sm = 0.f;
            for (int i = 0; i < 16; ++i) sm += red[i];
            s_lse = s_mx + __logf(sm);
        }
        __syncthreads();

        if (t == 0) {
            for (int s = tid; s < S_CTC; s += blockDim.x) {
                if (s == 0)      aold[s] = row[0] - s_lse;
                else if (s == 1) aold[s] = row[1] - s_lse;
                else             aold[s] = BIG_NEG;
            }
        } else {
            for (int s = tid; s < S_CTC; s += blockDim.x) {
                int ei = (s & 1) ? ((s + 1) >> 1) : 0;
                float e = row[ei] - s_lse;
                float xv = aold[s];
                xv = lae(xv, (s >= 1) ? aold[s - 1] : BIG_NEG);
                if ((s >= 3) && (s & 1)) xv = lae(xv, aold[s - 2]);
                anew[s] = xv + e;
            }
        }
        __syncthreads();
        if (t > 0) { float* tmp = aold; aold = anew; anew = tmp; }
    }
    if (tid == 0) nll[b] = -lae(aold[S_CTC - 1], aold[S_CTC - 2]);
}

// ------------------------- MAS forward (Q matrix) --------------------------
__global__ void aligner_mas_fwd(const float* __restrict__ lp, float* __restrict__ Q) {
    const int b = blockIdx.x;
    const int tid = threadIdx.x;
    const float* lpb = lp + (size_t)b * TFEATS * TTEXT;
    float* Qb = Q + (size_t)b * TFEATS * TTEXT;
    __shared__ float q[2][TTEXT];
    if (tid < TTEXT) {
        float v = (tid == 0) ? lpb[0] : BIG_NEG;
        q[0][tid] = v;
        Qb[tid] = v;
    }
    __syncthreads();
    int cur = 0;
    for (int t = 1; t < TFEATS; ++t) {
        if (tid < TTEXT) {
            float stay = q[cur][tid];
            float diag = (tid > 0) ? q[cur][tid - 1] : BIG_NEG;
            float v = fmaxf(stay, diag) + lpb[(size_t)t * TTEXT + tid];
            q[cur ^ 1][tid] = v;
            Qb[(size_t)t * TTEXT + tid] = v;
        }
        __syncthreads();
        cur ^= 1;
    }
}

// ------------------------- MAS backtrack -----------------------------------
__global__ void aligner_mas_bwd(const float* __restrict__ Q, int* __restrict__ A) {
    int b = threadIdx.x;
    if (b >= BB) return;
    const float* Qb = Q + (size_t)b * TFEATS * TTEXT;
    int* Ab = A + b * TFEATS;
    int a = TTEXT - 1;
    Ab[TFEATS - 1] = a;
    for (int t = TFEATS - 2; t >= 0; --t) {
        int an = a;
        int ia = (an - 1 > 0) ? (an - 1) : 0;
        int nv;
        if (an == 0) nv = 0;
        else nv = (Qb[(size_t)t * TTEXT + ia] >= Qb[(size_t)t * TTEXT + an]) ? ia : an;
        Ab[t] = nv;
        a = nv;
    }
}

// ------------------------- ds bincount + bin-loss partials -----------------
__global__ void aligner_ds_bin(const int* __restrict__ A, const float* __restrict__ lp,
                               float* __restrict__ ds, float* __restrict__ pbin) {
    const int b = blockIdx.x;
    const int tid = threadIdx.x;
    const int* Ab = A + b * TFEATS;
    const float* lpb = lp + (size_t)b * TFEATS * TTEXT;
    for (int n = tid; n < TTEXT; n += blockDim.x) {
        int cnt = 0;
        for (int t = 0; t < TFEATS; ++t) cnt += (Ab[t] == n) ? 1 : 0;
        ds[b * TTEXT + n] = (float)cnt;
    }
    __shared__ float red[512];
    float s = 0.f;
    for (int t = tid; t < TFEATS; t += blockDim.x)
        s += lpb[(size_t)t * TTEXT + Ab[t]];
    red[tid] = s;
    __syncthreads();
    for (int o = 256; o > 0; o >>= 1) {
        if (tid < o) red[tid] += red[tid + o];
        __syncthreads();
    }
    if (tid == 0) pbin[b] = red[0];
}

// ------------------------- final scalars -----------------------------------
__global__ void aligner_finalize(const float* __restrict__ nll, const float* __restrict__ pbin,
                                 float* __restrict__ out_bin, float* __restrict__ out_fsum) {
    if (threadIdx.x == 0 && blockIdx.x == 0) {
        float sb = 0.f, sf = 0.f;
        for (int b = 0; b < BB; ++b) {
            sb += pbin[b];
            sf += nll[b] / (float)TTEXT;
        }
        *out_bin  = -(sb / ((float)BB * (float)TFEATS));
        *out_fsum = sf / (float)BB;
    }
}

// ---------------------------------------------------------------------------
extern "C" void kernel_launch(void* const* d_in, const int* in_sizes, int n_in,
                              void* d_out, int out_size, void* d_ws, size_t ws_size,
                              hipStream_t stream) {
    const float* speech = (const float*)d_in[0];   // (16, 2000, 512)
    const float* txt    = (const float*)d_in[1];   // (16, 400, 256)
    const float* tw1 = (const float*)d_in[4];  const float* tb1 = (const float*)d_in[5];
    const float* tw2 = (const float*)d_in[6];  const float* tb2 = (const float*)d_in[7];
    const float* fw1 = (const float*)d_in[8];  const float* fb1 = (const float*)d_in[9];
    const float* fw2 = (const float*)d_in[10]; const float* fb2 = (const float*)d_in[11];
    const float* fw3 = (const float*)d_in[12]; const float* fb3 = (const float*)d_in[13];

    float* out = (float*)d_out;
    float* out_ds   = out;                              // 16*400
    float* out_bin  = out + (size_t)BB * TTEXT;         // 1
    float* out_fsum = out + (size_t)BB * TTEXT + 1;     // 1
    float* lp_out   = out + (size_t)BB * TTEXT + 2;     // 16*2000*400 (log_p_attn)

    uint8_t* wsb = (uint8_t*)d_ws;
    size_t off = 0;
    auto alloc = [&](size_t bytes) -> void* {
        void* p = wsb + off;
        off += (bytes + 255) & ~(size_t)255;
        return p;
    };
    _Float16* wt_t1 = (_Float16*)alloc((size_t)256 * 768 * 2);
    _Float16* wt_t2 = (_Float16*)alloc((size_t)256 * 256 * 2);
    _Float16* wt_f1 = (_Float16*)alloc((size_t)256 * 1536 * 2);
    _Float16* wt_f2 = (_Float16*)alloc((size_t)256 * 768 * 2);
    _Float16* wt_f3 = (_Float16*)alloc((size_t)256 * 256 * 2);
    _Float16* tbufA = (_Float16*)alloc((size_t)BB * TTEXT * NADIM * 2);
    _Float16* tbufB = (_Float16*)alloc((size_t)BB * TTEXT * NADIM * 2);
    _Float16* fbufA = (_Float16*)alloc((size_t)BB * TFEATS * NADIM * 2);
    _Float16* fbufB = (_Float16*)alloc((size_t)BB * TFEATS * NADIM * 2);
    float* fnorm = (float*)alloc((size_t)BB * TFEATS * 4);
    float* tnorm = (float*)alloc((size_t)BB * TTEXT * 4);
    float* prior = (float*)alloc((size_t)TFEATS * TTEXT * 4);
    float* Qmat  = (float*)alloc((size_t)BB * TFEATS * TTEXT * 4);
    int*   Apath = (int*)alloc((size_t)BB * TFEATS * 4);
    float* nll   = (float*)alloc(BB * 4);
    float* pbin  = (float*)alloc(BB * 4);
    (void)ws_size; (void)in_sizes; (void)n_in; (void)out_size;

    // 1) repack conv weights to f16 [O][k][c]
    auto repack = [&](const float* w, _Float16* o, int O, int C, int K) {
        int n = O * C * K;
        aligner_repack_w<<<(n + 255) / 256, 256, 0, stream>>>(w, o, O, C, K);
    };
    repack(tw1, wt_t1, 256, 256, 3);
    repack(tw2, wt_t2, 256, 256, 1);
    repack(fw1, wt_f1, 256, 512, 3);
    repack(fw2, wt_f2, 256, 256, 3);
    repack(fw3, wt_f3, 256, 256, 1);

    // 2) prior table
    aligner_prior<<<(TFEATS * TTEXT + 255) / 256, 256, 0, stream>>>(prior);

    // 3) conv chains (templated WMMA GEMMs, f16 activations)
    {
        dim3 blk(32);
        dim3 gt(TTEXT / 16, 4, BB);
        dim3 gf(TFEATS / 16, 4, BB);
        // text: relu(conv3 256->256) -> conv1 256->256
        aligner_conv_gemm<256, 3, float>   <<<gt, blk, 0, stream>>>(txt,   wt_t1, tb1, tbufA, TTEXT, 1);
        aligner_conv_gemm<256, 1, _Float16><<<gt, blk, 0, stream>>>(tbufA, wt_t2, tb2, tbufB, TTEXT, 0);
        // speech: relu(conv3 512->256) -> relu(conv3 256->256) -> conv1 256->256
        aligner_conv_gemm<512, 3, float>   <<<gf, blk, 0, stream>>>(speech, wt_f1, fb1, fbufA, TFEATS, 1);
        aligner_conv_gemm<256, 3, _Float16><<<gf, blk, 0, stream>>>(fbufA,  wt_f2, fb2, fbufB, TFEATS, 1);
        aligner_conv_gemm<256, 1, _Float16><<<gf, blk, 0, stream>>>(fbufB,  wt_f3, fb3, fbufA, TFEATS, 0);
    }

    // 4) squared norms
    aligner_row_norm<<<(BB * TFEATS + 255) / 256, 256, 0, stream>>>(fbufA, fnorm, BB * TFEATS, NADIM);
    aligner_row_norm<<<(BB * TTEXT + 255) / 256, 256, 0, stream>>>(tbufB, tnorm, BB * TTEXT, NADIM);

    // 5) distance scores (WMMA + TDM-staged B tile) into log_p_attn slot
    {
        dim3 g(TFEATS / 16, TTEXT / 80, BB), blk(32);
        aligner_score_gemm<<<g, blk, 0, stream>>>(fbufA, tbufB, fnorm, tnorm, lp_out);
    }

    // 6) log_softmax over text axis + prior, in place
    aligner_softmax_prior<<<(BB * TFEATS) / 8, 256, 0, stream>>>(lp_out, prior);

    // 7) CTC forward-sum loss
    aligner_ctc<<<BB, 512, 0, stream>>>(lp_out, nll);

    // 8) MAS Viterbi
    aligner_mas_fwd<<<BB, 512, 0, stream>>>(lp_out, Qmat);
    aligner_mas_bwd<<<1, 32, 0, stream>>>(Qmat, Apath);

    // 9) durations + bin-loss partials
    aligner_ds_bin<<<BB, 512, 0, stream>>>(Apath, lp_out, out_ds, pbin);

    // 10) final scalars
    aligner_finalize<<<1, 32, 0, stream>>>(nll, pbin, out_bin, out_fsum);
}
